// AXSLinearV2_74852690034819
// MI455X (gfx1250) — matrix operand order repaired and checked by reading, not compiled, not run on hardware
//
#include <hip/hip_runtime.h>
#include <hip/hip_bf16.h>

typedef __attribute__((ext_vector_type(16))) _Float16 v16h;
typedef __attribute__((ext_vector_type(8)))  float    v8f;
typedef int v4i __attribute__((vector_size(16)));

#define BM  128
#define BN  128
#define BKL 64   // K per stage (2 WMMA k-chunks)

#if defined(__gfx1250__) && \
    __has_builtin(__builtin_amdgcn_global_load_async_to_lds_b128) && \
    __has_builtin(__builtin_amdgcn_s_wait_asynccnt)
#define USE_ASYNC 1
#else
#define USE_ASYNC 0
#endif

#if USE_ASYNC
#define AS1(p) ((v4i __attribute__((address_space(1)))*)(unsigned long long)(p))
#define AS3(p) ((v4i __attribute__((address_space(3)))*)(unsigned int)(unsigned long long)(p))
#endif

// ---------------------------------------------------------------------------
// Kernel 1: NF5 levels (32) + midpoints (31) via erfinv (= ndtri up to scale).
// ---------------------------------------------------------------------------
__global__ void nf5_init(float* __restrict__ levels, float* __restrict__ mids) {
    __shared__ float L[32];
    int i = threadIdx.x;  // 32 threads
    float p = (i < 16) ? (0.015625f + (float)i * (0.484375f / 16.0f))
                       : (0.5f + (float)(i - 16) * (0.484375f / 15.0f));
    const float denom = 1.41421356237309515f * erfinvf(0.96875f);  // ndtri(0.984375)
    float v = (1.41421356237309515f * erfinvf(2.0f * p - 1.0f)) / denom;
    L[i] = v;
    levels[i] = v;
    __syncthreads();
    if (i < 31) mids[i] = 0.5f * (L[i] + L[i + 1]);
}

// ---------------------------------------------------------------------------
// Kernel 2: block-wise NF5 fake-quant (block=64 along last dim) -> fp16.
// One wave32 per quant block. TR=false: write q[r*C+k] packed (row-major).
// TR=true : write q[k*R+r] (transposed, for the GEMM's B operand).
// ---------------------------------------------------------------------------
template <bool TR>
__global__ __launch_bounds__(256) void nf5_quant(
    const float* __restrict__ x, _Float16* __restrict__ q,
    const float* __restrict__ levels, const float* __restrict__ mids,
    unsigned int nblocks, int cshift /* log2(C/64) */, int R, int C) {
    __shared__ float sl[32];
    if (threadIdx.x < 32) sl[threadIdx.x] = levels[threadIdx.x];
    float md[31];
#pragma unroll
    for (int j = 0; j < 31; ++j) md[j] = mids[j];
    __syncthreads();

    const int lane = threadIdx.x & 31;
    unsigned int blk = blockIdx.x * 8u + (threadIdx.x >> 5);
    if (blk >= nblocks) return;

    const float2 v2 = ((const float2*)x)[(size_t)blk * 32 + lane];
    float a0 = fabsf(v2.x), a1 = fabsf(v2.y);
    float m1 = fmaxf(a0, a1);
    float m2 = fminf(a0, a1);
#pragma unroll
    for (int off = 16; off; off >>= 1) {
        float o1 = __shfl_xor(m1, off);
        float o2 = __shfl_xor(m2, off);
        float hi = fmaxf(m1, o1);
        float lo = fminf(m1, o1);
        m2 = fmaxf(lo, fmaxf(m2, o2));
        m1 = hi;
    }
    // quantile(0.999, n=64): idx 62.937 -> lerp(sorted[62], sorted[63])
    const float frac = 0.999f * 63.0f - 62.0f;
    float scale = fmaxf(m2 + frac * (m1 - m2), 1e-8f);
    float inv = 1.0f / scale;

    float out[2];
    float vin[2] = {v2.x, v2.y};
#pragma unroll
    for (int e = 0; e < 2; ++e) {
        float nrm = fminf(fmaxf(vin[e] * inv, -1.0f), 1.0f);
        int idx = 0;
#pragma unroll
        for (int j = 0; j < 31; ++j) idx += (nrm > md[j]) ? 1 : 0;
        out[e] = sl[idx] * scale;  // fold block scale into the fp16 value
    }
    if (TR) {
        unsigned int r  = blk >> cshift;                 // row in [R)
        unsigned int kb = blk & ((1u << cshift) - 1u);   // 64-block within row
        unsigned int k  = kb * 64u + lane * 2u;
        q[(size_t)k * R + r]       = (_Float16)out[0];
        q[(size_t)(k + 1) * R + r] = (_Float16)out[1];
    } else {
        union { _Float16 h[2]; unsigned int u; } pk;
        pk.h[0] = (_Float16)out[0];
        pk.h[1] = (_Float16)out[1];
        ((unsigned int*)q)[(size_t)blk * 32 + lane] = pk.u;
    }
}

// ---------------------------------------------------------------------------
// Kernel 3: WMMA GEMM  C[m,n] = sum_k A[m,k]*Bt[k,n] + bias[n]
// A = qx [M,K] fp16, Bt = qw^T [K,N] fp16, C fp32 [M,N].
// 128x128 tile, 8 waves x (32x64), BK=64, double-buffered LDS fed by
// GLOBAL_LOAD_ASYNC_TO_LDS_B128 (ASYNCcnt) when available.
// ---------------------------------------------------------------------------
__global__ __launch_bounds__(256) void nf5_gemm(
    const _Float16* __restrict__ A, const _Float16* __restrict__ Bt,
    const float* __restrict__ bias, float* __restrict__ C,
    int M, int N, int K) {
#if USE_ASYNC
    __shared__ alignas(16) _Float16 As[2][BM * BKL];  // [m][k]
    __shared__ alignas(16) _Float16 Bs[2][BKL * BN];  // [k][n]
#else
    __shared__ alignas(16) _Float16 As[1][BM * BKL];
    __shared__ alignas(16) _Float16 Bs[1][BKL * BN];
#endif

    const int t    = threadIdx.x;
    const int wave = t >> 5;
    const int lane = t & 31;
    const int mw = (wave & 3) * 32;
    const int nw = (wave >> 2) * 64;
    const int mBase = blockIdx.y * BM;
    const int nBase = blockIdx.x * BN;

    v8f acc[2][4];
#pragma unroll
    for (int i = 0; i < 2; ++i)
#pragma unroll
        for (int j = 0; j < 4; ++j) acc[i][j] = v8f{};

    // staging roles: A: 2 thr/row, 32 halfs each; B: 4 thr/row, 32 halfs each
    const int arowS = t >> 1, akoff = (t & 1) * 32;
    const int brow  = t >> 2, bnoff = (t & 3) * 32;
    const _Float16* aG = A + (size_t)(mBase + arowS) * K + akoff;   // + k0
    const _Float16* bG = Bt + (size_t)brow * N + nBase + bnoff;     // + k0*N
    _Float16* aL0 = &As[0][arowS * BKL + akoff];
    _Float16* bL0 = &Bs[0][brow * BN + bnoff];

    // fragment addressing (ISA 16-bit A / B register layouts)
    const int arow = lane & 15;
    const int ksel = (lane >> 4) * 8;

#if USE_ASYNC
    _Float16* aL1 = &As[1][arowS * BKL + akoff];
    _Float16* bL1 = &Bs[1][brow * BN + bnoff];
    // prologue: stage k0 = 0 into buffer 0
    {
        const _Float16* ag = aG;
        const _Float16* bg = bG;
        __builtin_amdgcn_global_load_async_to_lds_b128(AS1(ag), AS3(aL0), 0, 0);
        __builtin_amdgcn_global_load_async_to_lds_b128(AS1(ag), AS3(aL0), 16, 0);
        __builtin_amdgcn_global_load_async_to_lds_b128(AS1(ag), AS3(aL0), 32, 0);
        __builtin_amdgcn_global_load_async_to_lds_b128(AS1(ag), AS3(aL0), 48, 0);
        __builtin_amdgcn_global_load_async_to_lds_b128(AS1(bg), AS3(bL0), 0, 0);
        __builtin_amdgcn_global_load_async_to_lds_b128(AS1(bg), AS3(bL0), 16, 0);
        __builtin_amdgcn_global_load_async_to_lds_b128(AS1(bg), AS3(bL0), 32, 0);
        __builtin_amdgcn_global_load_async_to_lds_b128(AS1(bg), AS3(bL0), 48, 0);
    }
    int b = 0;
#else
    const int b = 0;
#endif

    for (int k0 = 0; k0 < K; k0 += BKL) {
#if USE_ASYNC
        __builtin_amdgcn_s_wait_asynccnt(0);  // current buffer resident (own ops)
        __syncthreads();                      // ... for all waves
        int k1 = k0 + BKL;
        if (k1 < K) {                         // overlap DMA of next tile
            const _Float16* ag = aG + k1;
            const _Float16* bg = bG + (size_t)k1 * N;
            _Float16* al = b ? aL0 : aL1;
            _Float16* bl = b ? bL0 : bL1;
            __builtin_amdgcn_global_load_async_to_lds_b128(AS1(ag), AS3(al), 0, 0);
            __builtin_amdgcn_global_load_async_to_lds_b128(AS1(ag), AS3(al), 16, 0);
            __builtin_amdgcn_global_load_async_to_lds_b128(AS1(ag), AS3(al), 32, 0);
            __builtin_amdgcn_global_load_async_to_lds_b128(AS1(ag), AS3(al), 48, 0);
            __builtin_amdgcn_global_load_async_to_lds_b128(AS1(bg), AS3(bl), 0, 0);
            __builtin_amdgcn_global_load_async_to_lds_b128(AS1(bg), AS3(bl), 16, 0);
            __builtin_amdgcn_global_load_async_to_lds_b128(AS1(bg), AS3(bl), 32, 0);
            __builtin_amdgcn_global_load_async_to_lds_b128(AS1(bg), AS3(bl), 48, 0);
        }
#else
        const _Float16* ag = aG + k0;
        const _Float16* bg = bG + (size_t)k0 * N;
        uint4 ar0 = *(const uint4*)(ag);
        uint4 ar1 = *(const uint4*)(ag + 8);
        uint4 ar2 = *(const uint4*)(ag + 16);
        uint4 ar3 = *(const uint4*)(ag + 24);
        uint4 br0 = *(const uint4*)(bg);
        uint4 br1 = *(const uint4*)(bg + 8);
        uint4 br2 = *(const uint4*)(bg + 16);
        uint4 br3 = *(const uint4*)(bg + 24);
        __syncthreads();
        *(uint4*)(aL0)      = ar0;
        *(uint4*)(aL0 + 8)  = ar1;
        *(uint4*)(aL0 + 16) = ar2;
        *(uint4*)(aL0 + 24) = ar3;
        *(uint4*)(bL0)      = br0;
        *(uint4*)(bL0 + 8)  = br1;
        *(uint4*)(bL0 + 16) = br2;
        *(uint4*)(bL0 + 24) = br3;
        __syncthreads();
#endif
        // ---- compute: 2 k-chunks x (2 A-frags x 4 B-frags) ----
#pragma unroll
        for (int c = 0; c < 2; ++c) {
            v16h af[2], bf[4];
#pragma unroll
            for (int i = 0; i < 2; ++i) {
                const _Float16* p = &As[b][(mw + i * 16 + arow) * BKL + c * 32 + ksel];
                union { uint4 u[2]; v16h h; } f;
                f.u[0] = *(const uint4*)(p);
                f.u[1] = *(const uint4*)(p + 16);
                af[i] = f.h;
            }
#pragma unroll
            for (int j = 0; j < 4; ++j) {
                const _Float16* p = &Bs[b][(c * 32 + ksel) * BN + nw + j * 16 + arow];
                v16h h;
#pragma unroll
                for (int s = 0; s < 16; ++s) h[s] = p[(s < 8 ? s : s + 8) * BN];
                bf[j] = h;
            }
#pragma unroll
            for (int i = 0; i < 2; ++i)
#pragma unroll
                for (int j = 0; j < 4; ++j)
                    acc[i][j] = __builtin_amdgcn_wmma_f32_16x16x32_f16(
                        false, af[i], false, bf[j], (short)0, acc[i][j], false, false);
        }
#if USE_ASYNC
        b ^= 1;
#endif
    }

    // epilogue: lane%16 = N col; VGPR r + (lane>=16)*8 = M row
    const int col  = lane & 15;
    const int rsel = (lane >> 4) * 8;
#pragma unroll
    for (int j = 0; j < 4; ++j) {
        int n = nBase + nw + j * 16 + col;
        float bv = bias[n];
#pragma unroll
        for (int i = 0; i < 2; ++i) {
            int m0 = mBase + mw + i * 16 + rsel;
#pragma unroll
            for (int r = 0; r < 8; ++r)
                C[(size_t)(m0 + r) * N + n] = acc[i][j][r] + bv;
        }
    }
}

// ---------------------------------------------------------------------------
// Launch. Workspace: [levels 32f][mids 31f] pad 512B, qx fp16 (M*K),
// qw^T fp16 (K*N)  (~96 MB total).
// ---------------------------------------------------------------------------
extern "C" void kernel_launch(void* const* d_in, const int* in_sizes, int n_in,
                              void* d_out, int out_size, void* d_ws, size_t ws_size,
                              hipStream_t stream) {
    const float* x    = (const float*)d_in[0];  // [4,2048,4096]
    const float* w    = (const float*)d_in[1];  // [4096,4096]
    const float* bias = (const float*)d_in[2];  // [4096]
    float* out = (float*)d_out;

    const int N = in_sizes[2];
    const int K = in_sizes[1] / N;
    const int M = in_sizes[0] / K;

    float* levels = (float*)d_ws;
    float* mids   = levels + 32;
    _Float16* qx  = (_Float16*)((char*)d_ws + 512);
    _Float16* qwT = qx + (size_t)M * K;

    nf5_init<<<1, 32, 0, stream>>>(levels, mids);

    const int cshift = __builtin_ctz((unsigned)(K / 64));
    unsigned int nbx = (unsigned int)((size_t)M * K / 64);
    unsigned int nbw = (unsigned int)((size_t)N * K / 64);
    nf5_quant<false><<<(nbx + 7) / 8, 256, 0, stream>>>(x, qx, levels, mids, nbx, cshift, M, K);
    nf5_quant<true ><<<(nbw + 7) / 8, 256, 0, stream>>>(w, qwT, levels, mids, nbw, cshift, N, K);

    dim3 grid(N / BN, M / BM);
    nf5_gemm<<<grid, 256, 0, stream>>>(qx, qwT, bias, out, M, N, K);
}
